// SelfAttention_62826781606274
// MI455X (gfx1250) — matrix operand order repaired and checked by reading, not compiled, not run on hardware
//
#include <hip/hip_runtime.h>
#include <hip/hip_bf16.h>

// Self-attention, B=4, S=2048, D=1024, fp32 in/out.
// Pipeline: (1) QKV bf16-WMMA GEMM (+bias) -> (2) q@k^T bf16-WMMA GEMM with
// causal block skipping -> (3) row softmax -> (4) P@V bf16-WMMA GEMM with
// truncated K loop. Intermediates (80MB) live in d_ws and fit the 192MB L2.
// Kernels 2 and 4 use double-buffered GLOBAL_LOAD_ASYNC_TO_LDS_B128 staging
// (ASYNCcnt-tracked) so the WMMA pipe overlaps with memory->LDS traffic.

#define BATCH 4
#define SEQ   2048
#define DIM   1024

typedef __attribute__((ext_vector_type(16))) __bf16 v16bf;
typedef __attribute__((ext_vector_type(8)))  __bf16 v8bf;
typedef __attribute__((ext_vector_type(8)))  float  v8f;

union BF16x16 { v16bf v; v8bf h[2]; };

// ---- CDNA5 async-copy helpers ----------------------------------------------

// LDS byte offset of a generic pointer that points into LDS (addrspace 3).
__device__ __forceinline__ uint32_t lds_off_u32(const void* p) {
  return (uint32_t)(uintptr_t)(const __attribute__((address_space(3))) void*)p;
}

// One 16-byte memory->LDS async copy per lane (tracked with ASYNCcnt).
__device__ __forceinline__ void async_ld16(uint32_t lds_off, const void* gptr) {
  asm volatile("global_load_async_to_lds_b128 %0, %1, off"
               :: "v"(lds_off), "v"(gptr) : "memory");
}

#define WAIT_ASYNCCNT(imm) asm volatile("s_wait_asynccnt " #imm ::: "memory")

// Copy one 128x32 bf16 tile (row-major, leading dim ld, k-offset k0) into LDS
// asynchronously. 512 x 16B transfers / 256 threads = 2 async ops per thread.
__device__ __forceinline__ void stage_copy_async(__bf16* dst, const __bf16* src,
                                                 size_t ld, int k0, int tid) {
#pragma unroll
  for (int t = 0; t < 2; ++t) {
    int v = tid + t * 256;
    int r = v >> 2, c8 = (v & 3) * 8;
    async_ld16(lds_off_u32(dst + r * 32 + c8),
               src + (size_t)r * ld + k0 + c8);
  }
}

// ---- WMMA helpers (CDNA5: V_WMMA_F32_16X16X32_BF16, wave32) ----------------

__device__ __forceinline__ v8f wmma_bf16(v16bf a, v16bf b, v8f c) {
  // 8 args: (neg_a, A, neg_b, B, c_mod, C, reuse_a, reuse_b)
  return __builtin_amdgcn_wmma_f32_16x16x32_bf16(false, a, false, b,
                                                 (short)0, c, false, false);
}

// A tile is [16 rows][32 K] bf16 row-major in LDS (ld = 32).
// lanes 0-15 -> M = lane, K = 0..15;
// lanes 16-31 -> M = lane-16, K = 16..31.
__device__ __forceinline__ v16bf load_a_frag(const __bf16* base, int lane) {
  int r  = lane & 15;
  int kh = lane >> 4;               // 0 or 1
  const __bf16* p = base + r * 32 + kh * 16;
  BF16x16 f;
  f.h[0] = *(const v8bf*)(p);       // K = kh*16 + 0..7
  f.h[1] = *(const v8bf*)(p + 8);   // K = kh*16 + 8..15
  return f.v;
}

// B tile stored K-contiguous per column: [16 cols][32 K] bf16 (ld = 32).
// lanes 0-15 -> N = lane, K = 0..15; lanes 16-31 -> N = lane-16, K = 16..31.
__device__ __forceinline__ v16bf load_b_frag(const __bf16* base, int lane) {
  int n  = lane & 15;
  int kb = lane >> 4;
  return *(const v16bf*)(base + n * 32 + kb * 16);
}

// ---- Kernel 1: QKV projection  C = X @ W + b  (fp32 in, bf16 out) ----------
// grid: (8192/128, 1024/128, 3)   block: 256 (8 waves)
// Staging converts fp32->bf16 on the fly, so it stays on the VALU path.

__global__ __launch_bounds__(256)
void qkv_gemm(const float* __restrict__ x,
              const float* __restrict__ Wq, const float* __restrict__ Wk,
              const float* __restrict__ Wv,
              const float* __restrict__ bq, const float* __restrict__ bk,
              const float* __restrict__ bv,
              __bf16* __restrict__ qb, __bf16* __restrict__ kb,
              __bf16* __restrict__ vb) {
  const float* W; const float* bias; __bf16* outp;
  switch (blockIdx.z) {
    case 0:  W = Wq; bias = bq; outp = qb; break;
    case 1:  W = Wk; bias = bk; outp = kb; break;
    default: W = Wv; bias = bv; outp = vb; break;
  }
  const int row0 = blockIdx.x * 128;          // over B*S = 8192 rows
  const int n0   = blockIdx.y * 128;

  __shared__ __align__(16) __bf16 As[128 * 32];
  __shared__ __align__(16) __bf16 Bs[128 * 32];   // [n][k] transposed

  const int tid  = threadIdx.x;
  const int lane = tid & 31;
  const int w    = tid >> 5;
  const int wr   = w >> 1;                    // 0..3 (rows 32*wr)
  const int wc   = w & 1;                     // 0..1 (cols 64*wc)

  v8f acc[2][4] = {};

  for (int k0 = 0; k0 < DIM; k0 += 32) {
    // Stage A: convert fp32 x tile -> bf16 LDS.  1024 float4 loads / 256 thr.
    for (int i = tid; i < 1024; i += 256) {
      int r = i >> 3, c4 = (i & 7) * 4;
      float4 f = *(const float4*)(x + (size_t)(row0 + r) * DIM + k0 + c4);
      __bf16* d = &As[r * 32 + c4];
      d[0] = (__bf16)f.x; d[1] = (__bf16)f.y;
      d[2] = (__bf16)f.z; d[3] = (__bf16)f.w;
    }
    // Stage B: transpose fp32 W[k][n] tile -> bf16 Bs[n][k].
    for (int i = tid; i < 1024; i += 256) {
      int kk = i >> 5, n4 = (i & 31) * 4;
      float4 f = *(const float4*)(W + (size_t)(k0 + kk) * DIM + n0 + n4);
      Bs[(n4 + 0) * 32 + kk] = (__bf16)f.x;
      Bs[(n4 + 1) * 32 + kk] = (__bf16)f.y;
      Bs[(n4 + 2) * 32 + kk] = (__bf16)f.z;
      Bs[(n4 + 3) * 32 + kk] = (__bf16)f.w;
    }
    __syncthreads();
    v16bf af0 = load_a_frag(&As[(32 * wr) * 32], lane);
    v16bf af1 = load_a_frag(&As[(32 * wr + 16) * 32], lane);
#pragma unroll
    for (int jt = 0; jt < 4; ++jt) {
      v16bf bf = load_b_frag(&Bs[(64 * wc + 16 * jt) * 32], lane);
      acc[0][jt] = wmma_bf16(af0, bf, acc[0][jt]);
      acc[1][jt] = wmma_bf16(af1, bf, acc[1][jt]);
    }
    __syncthreads();
  }

  const int n_l = lane & 15, kh = lane >> 4;
#pragma unroll
  for (int it = 0; it < 2; ++it)
#pragma unroll
    for (int jt = 0; jt < 4; ++jt) {
      int gn = n0 + 64 * wc + 16 * jt + n_l;
      float bv_ = bias[gn];
#pragma unroll
      for (int i = 0; i < 8; ++i) {
        int gm = row0 + 32 * wr + 16 * it + 8 * kh + i;
        outp[(size_t)gm * DIM + gn] = (__bf16)(acc[it][jt][i] + bv_);
      }
    }
}

// ---- Kernel 2: scores = q @ k^T  (bf16 in, bf16 raw scores out) ------------
// grid: (16, 16, 4); fully-masked blocks early-return.
// Double-buffered async memory->LDS staging for both Q and K tiles.

__global__ __launch_bounds__(256)
void scores_gemm(const __bf16* __restrict__ qb, const __bf16* __restrict__ kb,
                 __bf16* __restrict__ probs) {
  const int b    = blockIdx.z;
  const int row0 = blockIdx.x * 128;      // query rows
  const int n0   = blockIdx.y * 128;      // key cols
  if (n0 > row0 + 127) return;            // causal: whole block masked

  const __bf16* Q = qb + (size_t)b * SEQ * DIM + (size_t)row0 * DIM;
  const __bf16* K = kb + (size_t)b * SEQ * DIM + (size_t)n0 * DIM;
  __bf16* P = probs + (size_t)b * SEQ * SEQ;

  __shared__ __align__(16) __bf16 As[2][128 * 32];
  __shared__ __align__(16) __bf16 Bs[2][128 * 32];

  const int tid  = threadIdx.x;
  const int lane = tid & 31;
  const int w    = tid >> 5;
  const int wr   = w >> 1;
  const int wc   = w & 1;

  v8f acc[2][4] = {};

  // Prologue: async-fill buffer 0 (4 async ops per wave: 2 for A, 2 for B).
  stage_copy_async(As[0], Q, DIM, 0, tid);
  stage_copy_async(Bs[0], K, DIM, 0, tid);

  const int nk = DIM / 32;
  for (int i = 0; i < nk; ++i) {
    const int p = i & 1;
    if (i + 1 < nk) {
      stage_copy_async(As[p ^ 1], Q, DIM, (i + 1) * 32, tid);
      stage_copy_async(Bs[p ^ 1], K, DIM, (i + 1) * 32, tid);
      WAIT_ASYNCCNT(0x4);     // in-order: leaves the 4 just-issued in flight
    } else {
      WAIT_ASYNCCNT(0x0);
    }
    __syncthreads();          // fill barrier: buffer p resident for all waves
    v16bf af0 = load_a_frag(&As[p][(32 * wr) * 32], lane);
    v16bf af1 = load_a_frag(&As[p][(32 * wr + 16) * 32], lane);
#pragma unroll
    for (int jt = 0; jt < 4; ++jt) {
      v16bf bf = load_b_frag(&Bs[p][(64 * wc + 16 * jt) * 32], lane);
      acc[0][jt] = wmma_bf16(af0, bf, acc[0][jt]);
      acc[1][jt] = wmma_bf16(af1, bf, acc[1][jt]);
    }
    __syncthreads();          // drain barrier: p may be overwritten next iter
  }

  const int n_l = lane & 15, kh = lane >> 4;
#pragma unroll
  for (int it = 0; it < 2; ++it)
#pragma unroll
    for (int jt = 0; jt < 4; ++jt) {
      int gn = n0 + 64 * wc + 16 * jt + n_l;
#pragma unroll
      for (int i = 0; i < 8; ++i) {
        int gm = row0 + 32 * wr + 16 * it + 8 * kh + i;
        P[(size_t)gm * SEQ + gn] = (__bf16)acc[it][jt][i];
      }
    }
}

// ---- Kernel 3: causal row softmax in place (scale = 1/sqrt(1024)) ----------
// grid: (B*S) blocks of 256 threads; each thread owns 8 strided columns.

__global__ __launch_bounds__(256)
void softmax_rows(__bf16* __restrict__ probs) {
  const int g = blockIdx.x;
  const int b = g >> 11;
  const int i = g & (SEQ - 1);
  __bf16* row = probs + ((size_t)b * SEQ + i) * SEQ;
  const float scale = 0.03125f;  // 1/sqrt(1024)
  const int tid = threadIdx.x;

  float vals[8];
  float m = -__builtin_inff();
#pragma unroll
  for (int t = 0; t < 8; ++t) {
    int j = tid + t * 256;
    float s = (j <= i) ? (float)row[j] : -__builtin_inff();
    vals[t] = s;
    m = fmaxf(m, s);
  }
  __shared__ float red[256];
  red[tid] = m; __syncthreads();
  for (int off = 128; off > 0; off >>= 1) {
    if (tid < off) red[tid] = fmaxf(red[tid], red[tid + off]);
    __syncthreads();
  }
  m = red[0];
  __syncthreads();

  float sum = 0.f;
#pragma unroll
  for (int t = 0; t < 8; ++t) {
    int j = tid + t * 256;
    float e = (j <= i) ? __expf(scale * (vals[t] - m)) : 0.f;
    vals[t] = e;
    sum += e;
  }
  red[tid] = sum; __syncthreads();
  for (int off = 128; off > 0; off >>= 1) {
    if (tid < off) red[tid] += red[tid + off];
    __syncthreads();
  }
  float inv = 1.f / red[0];
#pragma unroll
  for (int t = 0; t < 8; ++t) {
    int j = tid + t * 256;
    row[j] = (__bf16)(vals[t] * inv);   // masked cols get exact 0
  }
}

// ---- Kernel 4: out = P @ V  (bf16 in, fp32 out), K loop truncated at diag --
// grid: (16, 8, 4)
// P tile staged with double-buffered async copies; V tile needs a transpose
// so it is staged via VALU inside the same issue phase (overlaps with async).

__global__ __launch_bounds__(256)
void pv_gemm(const __bf16* __restrict__ probs, const __bf16* __restrict__ vb,
             float* __restrict__ out) {
  const int b    = blockIdx.z;
  const int row0 = blockIdx.x * 128;
  const int n0   = blockIdx.y * 128;
  const __bf16* P = probs + (size_t)b * SEQ * SEQ + (size_t)row0 * SEQ;
  const __bf16* V = vb + (size_t)b * SEQ * DIM;
  float* O = out + (size_t)b * SEQ * DIM;

  __shared__ __align__(16) __bf16 As[2][128 * 32];
  __shared__ __align__(16) __bf16 Bs[2][128 * 32];

  const int tid  = threadIdx.x;
  const int lane = tid & 31;
  const int w    = tid >> 5;
  const int wr   = w >> 1;
  const int wc   = w & 1;

  v8f acc[2][4] = {};

  const int kend = (row0 + 128 < SEQ) ? (row0 + 128) : SEQ;  // P is 0 beyond
  const int nk = kend / 32;

  // Stage V[k][n] tile transposed -> Bs[n][k] (VALU path, needs shuffle).
  auto stage_v = [&](__bf16* dst, int k0) {
#pragma unroll
    for (int t = 0; t < 2; ++t) {
      int v = tid + t * 256;
      int kk = v >> 4, n8 = (v & 15) * 8;
      uint4 raw = *(const uint4*)(V + (size_t)(k0 + kk) * DIM + n0 + n8);
      const __bf16* h = (const __bf16*)&raw;
#pragma unroll
      for (int j = 0; j < 8; ++j) dst[(n8 + j) * 32 + kk] = h[j];
    }
  };

  // Prologue: fill buffer 0 (2 async ops per wave for P, VALU stage for V).
  stage_copy_async(As[0], P, SEQ, 0, tid);
  stage_v(Bs[0], 0);

  for (int i = 0; i < nk; ++i) {
    const int p = i & 1;
    if (i + 1 < nk) {
      stage_copy_async(As[p ^ 1], P, SEQ, (i + 1) * 32, tid);
      stage_v(Bs[p ^ 1], (i + 1) * 32);
      WAIT_ASYNCCNT(0x2);
    } else {
      WAIT_ASYNCCNT(0x0);
    }
    __syncthreads();          // also waits DScnt for the V ds_stores
    v16bf af0 = load_a_frag(&As[p][(32 * wr) * 32], lane);
    v16bf af1 = load_a_frag(&As[p][(32 * wr + 16) * 32], lane);
#pragma unroll
    for (int jt = 0; jt < 4; ++jt) {
      v16bf bf = load_b_frag(&Bs[p][(64 * wc + 16 * jt) * 32], lane);
      acc[0][jt] = wmma_bf16(af0, bf, acc[0][jt]);
      acc[1][jt] = wmma_bf16(af1, bf, acc[1][jt]);
    }
    __syncthreads();
  }

  const int n_l = lane & 15, kh = lane >> 4;
#pragma unroll
  for (int it = 0; it < 2; ++it)
#pragma unroll
    for (int jt = 0; jt < 4; ++jt) {
      int gn = n0 + 64 * wc + 16 * jt + n_l;
#pragma unroll
      for (int i = 0; i < 8; ++i) {
        int gm = row0 + 32 * wr + 16 * it + 8 * kh + i;
        O[(size_t)gm * DIM + gn] = acc[it][jt][i];
      }
    }
}

// ---- host launch ------------------------------------------------------------

extern "C" void kernel_launch(void* const* d_in, const int* in_sizes, int n_in,
                              void* d_out, int out_size, void* d_ws, size_t ws_size,
                              hipStream_t stream) {
  const float* x  = (const float*)d_in[0];
  const float* Wq = (const float*)d_in[1];
  const float* Wk = (const float*)d_in[2];
  const float* Wv = (const float*)d_in[3];
  const float* bq = (const float*)d_in[4];
  const float* bk = (const float*)d_in[5];
  const float* bv = (const float*)d_in[6];
  float* out = (float*)d_out;

  // Workspace layout (bf16): q | k | v | probs  = 16+16+16+32 = 80 MB.
  const size_t qkv_elems = (size_t)BATCH * SEQ * DIM;
  __bf16* qb    = (__bf16*)d_ws;
  __bf16* kb    = qb + qkv_elems;
  __bf16* vb    = kb + qkv_elems;
  __bf16* probs = vb + qkv_elems;        // BATCH*SEQ*SEQ bf16

  dim3 blk(256);
  qkv_gemm<<<dim3((BATCH * SEQ) / 128, DIM / 128, 3), blk, 0, stream>>>(
      x, Wq, Wk, Wv, bq, bk, bv, qb, kb, vb);
  scores_gemm<<<dim3(SEQ / 128, SEQ / 128, BATCH), blk, 0, stream>>>(qb, kb, probs);
  softmax_rows<<<dim3(BATCH * SEQ), blk, 0, stream>>>(probs);
  pv_gemm<<<dim3(SEQ / 128, DIM / 128, BATCH), blk, 0, stream>>>(probs, vb, out);
}
